// DGCNN_72430328480594
// MI455X (gfx1250) — matrix operand order repaired and checked by reading, not compiled, not run on hardware
//
#include <hip/hip_runtime.h>
#include <hip/hip_bf16.h>
#include <math.h>

// ---------------- problem constants ----------------
#define NG     128          // graphs
#define NPER   512          // nodes per graph
#define NN     (NG*NPER)    // 65536 nodes
#define NE     1048576      // edges
#define DNODE  64
#define DEDGE  32
#define H0S    96           // h0 row stride / layer-0 K
#define CS     100          // hcat row stride (16B aligned; 97 cols used)
#define TOT    97
#define KTOP   30
#define NC1    16
#define NC2    32
#define KW     5
#define PL     15           // (30-2)/2+1
#define CLEN   11           // PL - KW + 1
#define DENSE  (CLEN*NC2)   // 352
#define NOUT   128

typedef __attribute__((ext_vector_type(16))) __bf16 v16bf;
typedef __attribute__((ext_vector_type(8)))  float  v8f;

__device__ __forceinline__ void atomAddF(float* p, float v) {
    __hip_atomic_fetch_add(p, v, __ATOMIC_RELAXED, __HIP_MEMORY_SCOPE_AGENT);
}

// -------- h0 = [node_feat | zeros], degs = 1 --------
__global__ __launch_bounds__(256) void init_h0(const float* __restrict__ nf,
                                               float* __restrict__ h0,
                                               float* __restrict__ degs) {
    int t = blockIdx.x * blockDim.x + threadIdx.x;
    if (t >= NN * H0S) return;
    int r = t / H0S, c = t - r * H0S;
    h0[t] = (c < DNODE) ? nf[(size_t)r * DNODE + c] : 0.0f;
    if (c == 0) degs[r] = 1.0f;
}

// -------- e2n = segment_sum(edge_feat, dst); degs += 1 per edge --------
__global__ __launch_bounds__(256) void e2n_scatter(const float* __restrict__ ef,
                                                   const int* __restrict__ dst,
                                                   float* __restrict__ h0,
                                                   float* __restrict__ degs) {
    int t = blockIdx.x * blockDim.x + threadIdx.x;
    int e = t >> 3;                         // 8 float4-groups per edge (32 ch)
    if (e >= NE) return;
    int c = (t & 7) * 4;
    int d = dst[e];
    float4 v = *(const float4*)(ef + (size_t)e * DEDGE + c);
    float* p = h0 + (size_t)d * H0S + DNODE + c;
    atomAddF(p + 0, v.x); atomAddF(p + 1, v.y);
    atomAddF(p + 2, v.z); atomAddF(p + 3, v.w);
    if (c == 0) atomAddF(&degs[d], 1.0f);
}

// -------- agg = h (row copy, float4) --------
__global__ __launch_bounds__(256) void copy4(const float* __restrict__ in, int in_stride,
                                             float* __restrict__ outp, int out_stride,
                                             int ngroups) {
    int t = blockIdx.x * blockDim.x + threadIdx.x;
    int r = t / ngroups;
    if (r >= NN) return;
    int c = (t - r * ngroups) * 4;
    *(float4*)(outp + (size_t)r * out_stride + c) =
        *(const float4*)(in + (size_t)r * in_stride + c);
}

// -------- agg[dst] += h[src]  (float4 gather + FP atomics) --------
__global__ __launch_bounds__(256) void edge_scatter(const float* __restrict__ h, int h_stride,
                                                    const int* __restrict__ src,
                                                    const int* __restrict__ dst,
                                                    float* __restrict__ agg, int ngroups) {
    int t = blockIdx.x * blockDim.x + threadIdx.x;
    int e = t / ngroups;
    if (e >= NE) return;
    int c = (t - e * ngroups) * 4;
    int s = src[e], d = dst[e];
    float4 v = *(const float4*)(h + (size_t)s * h_stride + c);
    float* p = agg + (size_t)d * H0S + c;
    atomAddF(p + 0, v.x); atomAddF(p + 1, v.y);
    atomAddF(p + 2, v.z); atomAddF(p + 3, v.w);
}

// -------- pack W (Kdim x outdim, f32) into WMMA B-fragment bf16 layout --------
// flat index i = ((nt*nchunks + chunk)*32 + lane)*16 + j
//   lane = half*16 + l16 ; column = nt*16 + l16 ; K = chunk*32 + half*16 + j
// out-of-range columns (outdim=1 layer) packed as 0 -> no predication in GEMM.
__global__ __launch_bounds__(256) void pack_W(const float* __restrict__ W,
                                              __bf16* __restrict__ Bp,
                                              int Kdim, int outdim, int ntiles) {
    int nchunks = Kdim >> 5;
    int total = ntiles * nchunks * 32 * 16;
    for (int i = blockIdx.x * blockDim.x + threadIdx.x; i < total;
         i += blockDim.x * gridDim.x) {
        int j    = i & 15;
        int lane = (i >> 4) & 31;
        int rest = i >> 9;
        int chunk = rest % nchunks;
        int nt    = rest / nchunks;
        int col = nt * 16 + (lane & 15);
        int k   = chunk * 32 + (lane >> 4) * 16 + j;
        float v = (col < outdim) ? W[(size_t)k * outdim + col] : 0.0f;
        Bp[i] = (__bf16)v;
    }
}

// -------- h_next = tanh((agg @ W + b)/deg) via v_wmma_f32_16x16x32_bf16 --------
// One wave per 16x16 output tile. A assembled from f32 agg (b128 clause + cvt),
// B fetched as one aligned 32B vector per K-chunk from the pre-packed buffer.
__global__ __launch_bounds__(256) void gemm_tanh(const float* __restrict__ in, int in_stride,
                                                 const __bf16* __restrict__ Bp,
                                                 const float* __restrict__ bias,
                                                 const float* __restrict__ degs,
                                                 float* __restrict__ outp, int out_stride,
                                                 int Kdim, int outdim, int ntiles) {
    int wave = (blockIdx.x * blockDim.x + threadIdx.x) >> 5;
    int lane = threadIdx.x & 31;
    int mt = wave / ntiles;
    int nt = wave - mt * ntiles;
    if (mt >= NN / 16) return;               // wave-uniform: EXEC stays all-ones for WMMA
    int half = lane >> 4;
    int l16  = lane & 15;
    int row_a = mt * 16 + l16;               // A row fetched by this lane
    int col   = nt * 16 + l16;               // output column owned by this lane
    const float* arow = in + (size_t)row_a * in_stride;
    int nchunks = Kdim >> 5;
    const v16bf* bbase = (const v16bf*)Bp + ((size_t)nt * nchunks) * 32 + lane;

    v8f acc = {};
    for (int c = 0; c < nchunks; ++c) {
        int kc = c * 32;
        v16bf a;
        // A 16x32 bf16 layout: elem s -> K = half*16 + s (matches packed B)
        #pragma unroll
        for (int j = 0; j < 8; ++j) {
            a[j]     = (__bf16)arow[kc + half * 16 + j];
            a[8 + j] = (__bf16)arow[kc + half * 16 + 8 + j];
        }
        v16bf b = bbase[(size_t)c * 32];     // pre-packed fragment, no predication
        acc = __builtin_amdgcn_wmma_f32_16x16x32_bf16(false, a, false, b,
                                                      (short)0, acc, false, false);
    }
    if (col < outdim) {
        float bv = bias[col];
        #pragma unroll
        for (int r = 0; r < 8; ++r) {        // D: VGPR r -> M = r + 8*half
            int row = mt * 16 + half * 8 + r;
            float v = tanhf((acc[r] + bv) / degs[row]);
            outp[(size_t)row * out_stride + col] = v;
        }
    }
}

// -------- per-graph: top-k sort-pool, conv1, maxpool, conv2, dense --------
__global__ __launch_bounds__(256) void tail_kernel(const float* __restrict__ hcat,
                                                   const float* __restrict__ cw1,
                                                   const float* __restrict__ cb1,
                                                   const float* __restrict__ cw2,
                                                   const float* __restrict__ cb2,
                                                   const float* __restrict__ ow,
                                                   const float* __restrict__ ob,
                                                   float* __restrict__ out) {
    __shared__ float vals[NPER];
    __shared__ int   topidx[KTOP];
    __shared__ float red_v[256];
    __shared__ int   red_i[256];
    __shared__ float pooled[KTOP][TOT];
    __shared__ float c1buf[NC1][KTOP];
    __shared__ float p1[NC1][PL];
    __shared__ float c2buf[DENSE];

    int g = blockIdx.x;
    int t = threadIdx.x;
    const float* base = hcat + (size_t)g * NPER * CS;

    for (int i = t; i < NPER; i += 256) vals[i] = base[(size_t)i * CS + (TOT - 1)];
    __syncthreads();

    // iterative argmax x30, stable tie-break (lower index first) like jax.lax.top_k
    for (int k = 0; k < KTOP; ++k) {
        float bv = -3.402823466e38f; int bi = 0x7fffffff;
        for (int i = t; i < NPER; i += 256) {
            float v = vals[i];
            if (v > bv) { bv = v; bi = i; }
        }
        red_v[t] = bv; red_i[t] = bi;
        __syncthreads();
        for (int s = 128; s > 0; s >>= 1) {
            if (t < s) {
                float v2 = red_v[t + s]; int i2 = red_i[t + s];
                if (v2 > red_v[t] || (v2 == red_v[t] && i2 < red_i[t])) {
                    red_v[t] = v2; red_i[t] = i2;
                }
            }
            __syncthreads();
        }
        if (t == 0) { topidx[k] = red_i[0]; vals[red_i[0]] = -3.402823466e38f; }
        __syncthreads();
    }

    // gather pooled (30 x 97)
    for (int i = t; i < KTOP * TOT; i += 256) {
        int k = i / TOT, d = i - k * TOT;
        pooled[k][d] = base[(size_t)topidx[k] * CS + d];
    }
    __syncthreads();

    // conv1 (kernel width 1 over feature dim): c1[o][k] = relu(dot(pooled[k], cw1[o]) + cb1[o])
    for (int i = t; i < NC1 * KTOP; i += 256) {
        int o = i / KTOP, k = i - o * KTOP;
        float s = cb1[o];
        for (int d = 0; d < TOT; ++d) s += pooled[k][d] * cw1[o * TOT + d];
        c1buf[o][k] = fmaxf(s, 0.0f);
    }
    __syncthreads();

    // max-pool pairs
    for (int i = t; i < NC1 * PL; i += 256) {
        int o = i / PL, p = i - o * PL;
        p1[o][p] = fmaxf(c1buf[o][2 * p], c1buf[o][2 * p + 1]);
    }
    __syncthreads();

    // conv2 (C1->C2, width 5, valid)
    for (int i = t; i < NC2 * CLEN; i += 256) {
        int o = i / CLEN, x = i - o * CLEN;
        float s = cb2[o];
        for (int ci = 0; ci < NC1; ++ci)
            for (int j = 0; j < KW; ++j)
                s += p1[ci][x + j] * cw2[o * (NC1 * KW) + ci * KW + j];
        c2buf[o * CLEN + x] = fmaxf(s, 0.0f);
    }
    __syncthreads();

    // dense 352 -> 128 ; relu(relu(x)) == relu(x)
    for (int o = t; o < NOUT; o += 256) {
        float s = ob[o];
        for (int f = 0; f < DENSE; ++f) s += c2buf[f] * ow[(size_t)f * NOUT + o];
        out[(size_t)g * NOUT + o] = fmaxf(s, 0.0f);
    }
}

extern "C" void kernel_launch(void* const* d_in, const int* in_sizes, int n_in,
                              void* d_out, int out_size, void* d_ws, size_t ws_size,
                              hipStream_t stream) {
    const float* node_feat = (const float*)d_in[0];
    const float* edge_feat = (const float*)d_in[1];
    const int*   edge_src  = (const int*)d_in[2];
    const int*   edge_dst  = (const int*)d_in[3];
    const float* Wm[4] = { (const float*)d_in[4], (const float*)d_in[6],
                           (const float*)d_in[8], (const float*)d_in[10] };
    const float* bm[4] = { (const float*)d_in[5], (const float*)d_in[7],
                           (const float*)d_in[9], (const float*)d_in[11] };
    const float* cw1 = (const float*)d_in[12];
    const float* cb1 = (const float*)d_in[13];
    const float* cw2 = (const float*)d_in[14];
    const float* cb2 = (const float*)d_in[15];
    const float* ow  = (const float*)d_in[16];
    const float* ob  = (const float*)d_in[17];
    float* out = (float*)d_out;

    const int Kdim[4]   = { 96, 32, 32, 32 };
    const int outdim[4] = { 32, 32, 32, 1 };
    const int coloff[4] = { 0, 32, 64, 96 };
    const int ntl[4]    = { 2, 2, 2, 1 };

    // workspace layout (16B/64B-aligned offsets)
    char* ws = (char*)d_ws;
    size_t off = 0;
    float* h0   = (float*)(ws + off); off += (size_t)NN * H0S * 4;   // 25.2 MB
    float* agg  = (float*)(ws + off); off += (size_t)NN * H0S * 4;   // 25.2 MB
    float* hcat = (float*)(ws + off); off += (size_t)NN * CS * 4;    // 26.2 MB
    float* degs = (float*)(ws + off); off += (size_t)NN * 4;         // 256 KB
    __bf16* Bp[4];
    for (int l = 0; l < 4; ++l) {
        Bp[l] = (__bf16*)(ws + off);
        size_t elems = (size_t)ntl[l] * (Kdim[l] / 32) * 32 * 16;    // <= 3072
        off += ((elems * 2 + 63) / 64) * 64;                         // keep 64B aligned
    }

    // 0) pack weights into WMMA B-fragment layout (tiny)
    for (int l = 0; l < 4; ++l)
        pack_W<<<4, 256, 0, stream>>>(Wm[l], Bp[l], Kdim[l], outdim[l], ntl[l]);

    // 1) h0 = [node_feat | 0], degs = 1
    init_h0<<<(NN * H0S + 255) / 256, 256, 0, stream>>>(node_feat, h0, degs);
    // 2) e2n segment-sum + degree counts
    e2n_scatter<<<(NE * 8 + 255) / 256, 256, 0, stream>>>(edge_feat, edge_dst, h0, degs);

    for (int l = 0; l < 4; ++l) {
        const float* hin = (l == 0) ? h0 : (hcat + coloff[l - 1]);
        int hstride = (l == 0) ? H0S : CS;
        int ng = Kdim[l] / 4;
        // agg = h
        copy4<<<((size_t)NN * ng + 255) / 256, 256, 0, stream>>>(hin, hstride, agg, H0S, ng);
        // agg[dst] += h[src]
        edge_scatter<<<((size_t)NE * ng + 255) / 256, 256, 0, stream>>>(hin, hstride,
                                                                        edge_src, edge_dst,
                                                                        agg, ng);
        // h_next = tanh((agg @ W + b)/deg) -> hcat columns [coloff, coloff+outdim)
        size_t waves = (size_t)(NN / 16) * ntl[l];
        gemm_tanh<<<(waves * 32 + 255) / 256, 256, 0, stream>>>(agg, H0S, Bp[l], bm[l], degs,
                                                                hcat + coloff[l], CS,
                                                                Kdim[l], outdim[l], ntl[l]);
    }

    // 3) per-graph sort-pool + CNN head
    tail_kernel<<<NG, 256, 0, stream>>>(hcat, cw1, cb1, cw2, cb2, ow, ob, out);
}
